// CoarseMatching_39410619908298
// MI455X (gfx1250) — matrix operand order repaired and checked by reading, not compiled, not run on hardware
//
#include <hip/hip_runtime.h>

// ---------------- problem constants ----------------
#define N_      2
#define C_      256
#define H0_     60
#define W0_     80
#define H1_     60
#define W1_     80
#define L_      4800        // H0*W0
#define S_      4800        // H1*W1
#define THR_    0.2f
#define BORDER_ 2
#define SIM_SCALE 0.0390625f  // 1/(C * TEMP) = 1/(256*0.1)

// GEMM tiling: 32x32 output per wave (2x2 register tile of 16x16 WMMA tiles)
#define TILES_M (L_/32)                     // 150
#define TILES_S (S_/32)                     // 150
#define TILES_PER_N (TILES_M*TILES_S)       // 22500
#define TILES_TOTAL (N_*TILES_PER_N)        // 45000
#define WAVES_PER_BLOCK 8
#define GEMM_BLOCKS (TILES_TOTAL/WAVES_PER_BLOCK)  // 5625 (exact)

typedef __attribute__((ext_vector_type(16))) __bf16 v16bf;
typedef __attribute__((ext_vector_type(8)))  __bf16 v8bf;
typedef __attribute__((ext_vector_type(8)))  float  v8f;

// ---------------- pass 0: split f0 into bf16 hi/lo, same [n][l][c] layout ----
__global__ void __launch_bounds__(256)
k_split_f0(const float* __restrict__ f0, __bf16* __restrict__ hi,
           __bf16* __restrict__ lo) {
    size_t i = (size_t)blockIdx.x * 256 + threadIdx.x;   // exact: N*L*C = 9600*256
    float x = f0[i];
    __bf16 h = (__bf16)x;
    hi[i] = h;
    lo[i] = (__bf16)(x - (float)h);
}

// ---------------- pass 1: split f1 into bf16 hi/lo, same [n][s][c] layout ------
__global__ void __launch_bounds__(256)
k_split_tr_f1(const float* __restrict__ f1, __bf16* __restrict__ hi,
              __bf16* __restrict__ lo) {
    size_t i = (size_t)blockIdx.x * 256 + threadIdx.x;   // exact: N*C*S = 9600*256
    int s = (int)(i % S_);
    size_t r = i / S_;
    int c = (int)(r % C_);
    int n = (int)(r / C_);
    float x = f1[((size_t)n * S_ + s) * C_ + c];
    __bf16 h = (__bf16)x;
    hi[((size_t)n * S_ + s) * C_ + c] = h;
    lo[((size_t)n * S_ + s) * C_ + c] = (__bf16)(x - (float)h);
}

// ---------------- helper: load A fragment (ISA 16-bit A 16x32 layout) -------
__device__ __forceinline__ v16bf load_a_frag(const __bf16* p) {
    v8bf a0 = *(const v8bf*)(p);
    v8bf a1 = *(const v8bf*)(p + 16);
    return __builtin_shufflevector(a0, a1, 0,1,2,3,4,5,6,7,8,9,10,11,12,13,14,15);
}

// ---------------- pass 2: WMMA GEMM, sim = (f0 . f1^T) * SIM_SCALE ----------
// 32x32 output tile per wave (2x2 WMMA tiles): each A fragment is reused by
// 2 B tiles and vice versa -> 16 b128 loads per 12 WMMA per K-step.
// A: [n][l][c] bf16 (K-contiguous).  B: [n][s][c] bf16 (K-contiguous).
__global__ void __launch_bounds__(256)
k_gemm(const __bf16* __restrict__ Ahi, const __bf16* __restrict__ Alo,
       const __bf16* __restrict__ Bhi, const __bf16* __restrict__ Blo,
       float* __restrict__ sim) {
    const int lane = threadIdx.x & 31;
    const int wave = threadIdx.x >> 5;
    const int tile = blockIdx.x * WAVES_PER_BLOCK + wave; // always < TILES_TOTAL
    const int n  = tile / TILES_PER_N;
    const int r  = tile % TILES_PER_N;
    const int tm = r / TILES_S;
    const int ts = r % TILES_S;

    // A fragment addressing (ISA 16-bit A 16x32 layout):
    //   lanes 0-15: row M=lane, klo=0 ; lanes 16-31: row M=lane-16, klo=8
    //   elements 0..7 -> K = k0+klo+0..7 ; elements 8..15 -> K = k0+klo+16..23
    const int rowA = tm * 32 + (lane & 15);
    const int klo  = (lane >> 4) << 3;                 // 0 or 8
    const __bf16* pAh0 = Ahi + ((size_t)n * L_ + rowA) * C_ + klo;
    const __bf16* pAl0 = Alo + ((size_t)n * L_ + rowA) * C_ + klo;
    const __bf16* pAh1 = pAh0 + (size_t)16 * C_;
    const __bf16* pAl1 = pAl0 + (size_t)16 * C_;

    // B fragment addressing (same K order as A):
    //   lanes 0-15: col N=lane, klo=0 ; lanes 16-31: col N=lane-16, klo=8
    //   elements 0..7 -> K = k0+klo+0..7 ; elements 8..15 -> K = k0+klo+16..23
    const int colB = ts * 32 + (lane & 15);
    const __bf16* pBh = Bhi + ((size_t)n * S_ + colB) * C_ + klo;
    const __bf16* pBl = Blo + ((size_t)n * S_ + colB) * C_ + klo;

    v8f acc00 = {}, acc01 = {}, acc10 = {}, acc11 = {};
#pragma unroll
    for (int k0 = 0; k0 < C_; k0 += 32) {
        v16bf ah0 = load_a_frag(pAh0 + k0);
        v16bf al0 = load_a_frag(pAl0 + k0);
        v16bf ah1 = load_a_frag(pAh1 + k0);
        v16bf al1 = load_a_frag(pAl1 + k0);
        v16bf bh0 = load_a_frag(pBh + k0);
        v16bf bh1 = load_a_frag(pBh + (size_t)16 * C_ + k0);
        v16bf bl0 = load_a_frag(pBl + k0);
        v16bf bl1 = load_a_frag(pBl + (size_t)16 * C_ + k0);
        // bf16x3 split per tile: hi*hi + hi*lo + lo*hi (~f32 accuracy)
        acc00 = __builtin_amdgcn_wmma_f32_16x16x32_bf16(false, ah0, false, bh0, (short)0, acc00, false, false);
        acc00 = __builtin_amdgcn_wmma_f32_16x16x32_bf16(false, ah0, false, bl0, (short)0, acc00, false, false);
        acc00 = __builtin_amdgcn_wmma_f32_16x16x32_bf16(false, al0, false, bh0, (short)0, acc00, false, false);

        acc01 = __builtin_amdgcn_wmma_f32_16x16x32_bf16(false, ah0, false, bh1, (short)0, acc01, false, false);
        acc01 = __builtin_amdgcn_wmma_f32_16x16x32_bf16(false, ah0, false, bl1, (short)0, acc01, false, false);
        acc01 = __builtin_amdgcn_wmma_f32_16x16x32_bf16(false, al0, false, bh1, (short)0, acc01, false, false);

        acc10 = __builtin_amdgcn_wmma_f32_16x16x32_bf16(false, ah1, false, bh0, (short)0, acc10, false, false);
        acc10 = __builtin_amdgcn_wmma_f32_16x16x32_bf16(false, ah1, false, bl0, (short)0, acc10, false, false);
        acc10 = __builtin_amdgcn_wmma_f32_16x16x32_bf16(false, al1, false, bh0, (short)0, acc10, false, false);

        acc11 = __builtin_amdgcn_wmma_f32_16x16x32_bf16(false, ah1, false, bh1, (short)0, acc11, false, false);
        acc11 = __builtin_amdgcn_wmma_f32_16x16x32_bf16(false, ah1, false, bl1, (short)0, acc11, false, false);
        acc11 = __builtin_amdgcn_wmma_f32_16x16x32_bf16(false, al1, false, bh1, (short)0, acc11, false, false);
    }

    // C/D layout: lanes 0-15 N=lane, M=ri ; lanes 16-31 N=lane-16, M=ri+8
    const int nc = lane & 15;
    const int mb = (lane >> 4) << 3;                   // 0 or 8
    float* out = sim + ((size_t)n * L_ + (size_t)tm * 32) * S_ + (size_t)ts * 32 + nc;
#pragma unroll
    for (int ri = 0; ri < 8; ++ri) {
        out[(size_t)(mb + ri) * S_]           = acc00[ri] * SIM_SCALE;
        out[(size_t)(mb + ri) * S_ + 16]      = acc01[ri] * SIM_SCALE;
        out[(size_t)(16 + mb + ri) * S_]      = acc10[ri] * SIM_SCALE;
        out[(size_t)(16 + mb + ri) * S_ + 16] = acc11[ri] * SIM_SCALE;
    }
}

// ---------------- pass 3: per-row softmax stats (max, sumexp over s) --------
__global__ void __launch_bounds__(256)
k_row_stats(const float* __restrict__ sim, float* __restrict__ rmax,
            float* __restrict__ rsum) {
    __shared__ float red[256];
    const int row = blockIdx.x;                        // n*L + l
    const float* p = sim + (size_t)row * S_;
    const int tid = threadIdx.x;
    float m = -1e30f;
    for (int s = tid; s < S_; s += 256) m = fmaxf(m, p[s]);
    red[tid] = m; __syncthreads();
    for (int off = 128; off > 0; off >>= 1) {
        if (tid < off) red[tid] = fmaxf(red[tid], red[tid + off]);
        __syncthreads();
    }
    m = red[0]; __syncthreads();
    float sum = 0.f;
    for (int s = tid; s < S_; s += 256) sum += expf(p[s] - m);
    red[tid] = sum; __syncthreads();
    for (int off = 128; off > 0; off >>= 1) {
        if (tid < off) red[tid] += red[tid + off];
        __syncthreads();
    }
    if (tid == 0) { rmax[row] = m; rsum[row] = red[0]; }
}

// ---------------- pass 4: per-column softmax stats (online, over l) ---------
__global__ void __launch_bounds__(256)
k_col_stats(const float* __restrict__ sim, float* __restrict__ cmax,
            float* __restrict__ csum) {
    int idx = blockIdx.x * 256 + threadIdx.x;          // over N*S
    if (idx >= N_ * S_) return;
    int n = idx / S_, s = idx % S_;
    const float* p = sim + (size_t)n * L_ * S_ + s;
    float m = -1e30f, sum = 0.f;
    for (int l = 0; l < L_; ++l) {
        float x = p[(size_t)l * S_];
        if (x > m) { sum = sum * expf(m - x) + 1.0f; m = x; }
        else         sum += expf(x - m);
    }
    cmax[idx] = m; csum[idx] = sum;
}

// ---------------- pass 5: conf = softmax_l * softmax_s, in place ------------
__global__ void __launch_bounds__(256)
k_conf(float* __restrict__ sim, const float* __restrict__ rmax,
       const float* __restrict__ rsum, const float* __restrict__ cmax,
       const float* __restrict__ csum) {
    size_t i = (size_t)blockIdx.x * 256 + threadIdx.x; // exact: N*L*S = 180000*256
    size_t nl = i / S_;
    int s = (int)(i % S_);
    int n = (int)(nl / L_);
    float x = sim[i];
    float c = expf((x - rmax[nl]) + (x - cmax[(size_t)n * S_ + s]))
              / (rsum[nl] * csum[(size_t)n * S_ + s]);
    sim[i] = c;
}

// ---------------- pass 6: column max of conf (over l) -----------------------
__global__ void __launch_bounds__(256)
k_col_confmax(const float* __restrict__ conf, float* __restrict__ cmaxC) {
    int idx = blockIdx.x * 256 + threadIdx.x;          // over N*S
    if (idx >= N_ * S_) return;
    int n = idx / S_, s = idx % S_;
    const float* p = conf + (size_t)n * L_ * S_ + s;
    float m = -1e30f;
    for (int l = 0; l < L_; ++l) m = fmaxf(m, p[(size_t)l * S_]);
    cmaxC[idx] = m;
}

// ---------------- pass 7: mutual-NN + threshold + border, per row -----------
__global__ void __launch_bounds__(256)
k_match(const float* __restrict__ conf, const float* __restrict__ cmaxC,
        float* __restrict__ mask_v, float* __restrict__ j_ids,
        float* __restrict__ mconf) {
    __shared__ float redf[256];
    __shared__ int   redi[256];
    const int row = blockIdx.x;                        // n*L + l
    const int n = row / L_, l = row % L_;
    const int tid = threadIdx.x;
    const float* p = conf + (size_t)row * S_;

    // row max of conf (over ALL s, unmasked, per reference)
    float m = -1e30f;
    for (int s = tid; s < S_; s += 256) m = fmaxf(m, p[s]);
    redf[tid] = m; __syncthreads();
    for (int off = 128; off > 0; off >>= 1) {
        if (tid < off) redf[tid] = fmaxf(redf[tid], redf[tid + off]);
        __syncthreads();
    }
    m = redf[0]; __syncthreads();

    const int i0 = l / W0_, j0 = l % W0_;
    const bool okl = (i0 >= BORDER_) && (i0 < H0_ - BORDER_) &&
                     (j0 >= BORDER_) && (j0 < W0_ - BORDER_);
    int best = S_;
    if (okl) {
        for (int s = tid; s < S_; s += 256) {
            float v = p[s];
            int i1 = s / W1_, j1 = s % W1_;
            bool ok = (v > THR_) &&
                      (i1 >= BORDER_) && (i1 < H1_ - BORDER_) &&
                      (j1 >= BORDER_) && (j1 < W1_ - BORDER_) &&
                      (v == m) && (v == cmaxC[(size_t)n * S_ + s]);
            if (ok && s < best) best = s;
        }
    }
    redi[tid] = best; __syncthreads();
    for (int off = 128; off > 0; off >>= 1) {
        if (tid < off) redi[tid] = min(redi[tid], redi[tid + off]);
        __syncthreads();
    }
    if (tid == 0) {
        int b = redi[0];
        bool found = (b < S_);
        mask_v[row] = found ? 1.0f : 0.0f;
        j_ids[row]  = found ? (float)b : 0.0f;
        mconf[row]  = found ? p[b] : 0.0f;
    }
}

// ---------------- launcher ---------------------------------------------------
extern "C" void kernel_launch(void* const* d_in, const int* in_sizes, int n_in,
                              void* d_out, int out_size, void* d_ws, size_t ws_size,
                              hipStream_t stream) {
    const float* f0 = (const float*)d_in[0];   // [N,L,C]
    const float* f1 = (const float*)d_in[1];   // [N,S,C]

    float* conf = (float*)d_out;                           // N*L*S
    float* mask_v = conf + (size_t)N_ * L_ * S_;           // N*L
    float* j_ids  = mask_v + (size_t)N_ * L_;              // N*L
    float* mconf  = j_ids  + (size_t)N_ * L_;              // N*L

    // workspace layout (all 256B aligned)
    char* ws = (char*)d_ws;
    const size_t szF  = (size_t)N_ * L_ * C_ * sizeof(__bf16);  // 4,915,200 B
    __bf16* f0hi = (__bf16*)(ws);
    __bf16* f0lo = (__bf16*)(ws + szF);
    __bf16* f1hi = (__bf16*)(ws + 2 * szF);
    __bf16* f1lo = (__bf16*)(ws + 3 * szF);
    float*  rmax = (float*)(ws + 4 * szF);
    float*  rsum = rmax + (size_t)N_ * L_;
    float*  cmax = rsum + (size_t)N_ * L_;
    float*  csum = cmax + (size_t)N_ * S_;
    float*  cmaxC = csum + (size_t)N_ * S_;

    const int elemsF = N_ * L_ * C_;                        // 2,457,600 = 9600*256

    k_split_f0   <<<elemsF / 256, 256, 0, stream>>>(f0, f0hi, f0lo);
    k_split_tr_f1<<<elemsF / 256, 256, 0, stream>>>(f1, f1hi, f1lo);
    k_gemm       <<<GEMM_BLOCKS, 256, 0, stream>>>(f0hi, f0lo, f1hi, f1lo, conf);
    k_row_stats  <<<N_ * L_, 256, 0, stream>>>(conf, rmax, rsum);
    k_col_stats  <<<(N_ * S_ + 255) / 256, 256, 0, stream>>>(conf, cmax, csum);
    k_conf       <<<(int)(((size_t)N_ * L_ * S_) / 256), 256, 0, stream>>>(conf, rmax, rsum, cmax, csum);
    k_col_confmax<<<(N_ * S_ + 255) / 256, 256, 0, stream>>>(conf, cmaxC);
    k_match      <<<N_ * L_, 256, 0, stream>>>(conf, cmaxC, mask_v, j_ids, mconf);
}
